// HyperEncoder_26216480375347
// MI455X (gfx1250) — matrix operand order repaired and checked by reading, not compiled, not run on hardware
//
#include <hip/hip_runtime.h>
#include <math.h>

#define N_NODES 100000
#define N_EDGES 20000
#define NNZV    500000
#define DD      128
#define NLAYERS 4
#define ALPHA_C 0.1f
#define LAMDA_C 0.5f

typedef __bf16 bf16_t;
typedef __attribute__((ext_vector_type(16))) __bf16 v16bf;
typedef __attribute__((ext_vector_type(8)))  float  v8f;

// ---------- ordered-float <-> uint keys for atomic segment-max ----------
__device__ __forceinline__ unsigned f2key(float f) {
  unsigned u = __float_as_uint(f);
  return (u & 0x80000000u) ? ~u : (u | 0x80000000u);
}
__device__ __forceinline__ float key2f(unsigned u) {
  return (u & 0x80000000u) ? __uint_as_float(u & 0x7fffffffu)
                           : __uint_as_float(~u);
}

// ---------- zero fill ----------
__global__ void fill0_kernel(float* __restrict__ p, int n) {
  int i = blockIdx.x * 256 + threadIdx.x;
  if (i < n) p[i] = 0.0f;
}

// ---------- transpose + bf16-convert all 9 weight matrices ----------
// WT[mat][n*128 + k] = (bf16) W[mat][k*128 + n]
__global__ void wtrans_kernel(const float* __restrict__ W0,
                              const float* __restrict__ Wn,
                              const float* __restrict__ We,
                              bf16_t* __restrict__ WT) {
  int id = blockIdx.x * 256 + threadIdx.x;
  if (id >= 9 * DD * DD) return;
  int mat = id >> 14;
  int r   = id & 16383;
  int n = r >> 7, k = r & 127;
  const float* src = (mat == 0) ? W0
                   : (mat <= 4) ? (Wn + (size_t)(mat - 1) * DD * DD)
                                : (We + (size_t)(mat - 5) * DD * DD);
  WT[((size_t)mat << 14) + (size_t)n * DD + k] = (bf16_t)src[(size_t)k * DD + n];
}

// ---------- degree kernels ----------
__global__ void deg_count_kernel(const int* __restrict__ nodes,
                                 const int* __restrict__ edges,
                                 float* dV, float* cntE, int nnz) {
  int k = blockIdx.x * 256 + threadIdx.x;
  if (k >= nnz) return;
  atomicAdd(dV + nodes[k], 1.0f);
  atomicAdd(cntE + edges[k], 1.0f);
}

__global__ void deg_sum_kernel(const int* __restrict__ nodes,
                               const int* __restrict__ edges,
                               const float* __restrict__ dV,
                               float* dEsum, int nnz) {
  int k = blockIdx.x * 256 + threadIdx.x;
  if (k >= nnz) return;
  atomicAdd(dEsum + edges[k], dV[nodes[k]]);
}

__global__ void deg_final_kernel(const float* __restrict__ dV,
                                 const float* __restrict__ cntE,
                                 const float* __restrict__ dEsum,
                                 float* degV, float* degE, int n, int e) {
  int i = blockIdx.x * 256 + threadIdx.x;
  if (i < n) { float d = dV[i]; degV[i] = d > 0.0f ? rsqrtf(d) : 1.0f; }
  if (i < e) {
    float d = dEsum[i] / fmaxf(cntE[i], 1.0f);
    degE[i] = d > 0.0f ? rsqrtf(d) : 1.0f;
  }
}

// ---------- per-row attention projections: one wave (32 lanes) per row ----------
__global__ __launch_bounds__(256) void proj_kernel(
    const float* __restrict__ xcur, const float* __restrict__ ecur,
    const float* __restrict__ attn_w, int layer,
    float* sx1, float* sx2, float* se1, float* se2, int n, int e) {
  int row  = blockIdx.x * 8 + (threadIdx.x >> 5);
  int lane = threadIdx.x & 31;
  if (row >= n + e) return;
  const float* aw = attn_w + (size_t)layer * 2 * DD;
  float4 lo = *(const float4*)(aw + lane * 4);
  float4 hi = *(const float4*)(aw + DD + lane * 4);
  const float* rp = (row < n) ? (xcur + (size_t)row * DD)
                              : (ecur + (size_t)(row - n) * DD);
  float4 rv = *(const float4*)(rp + lane * 4);
  float s1 = rv.x * lo.x + rv.y * lo.y + rv.z * lo.z + rv.w * lo.w;
  float s2 = rv.x * hi.x + rv.y * hi.y + rv.z * hi.z + rv.w * hi.w;
#pragma unroll
  for (int off = 16; off > 0; off >>= 1) {
    s1 += __shfl_down(s1, off, 32);
    s2 += __shfl_down(s2, off, 32);
  }
  if (lane == 0) {
    if (row < n) { sx1[row] = s1; sx2[row] = s2; }
    else         { se2[row - n] = s1; se1[row - n] = s2; }
  }
}

// ---------- per-nnz scores + segment max ----------
__global__ void scores_kernel(
    const int* __restrict__ nodes, const int* __restrict__ edges,
    const float* __restrict__ sx1, const float* __restrict__ se1,
    const float* __restrict__ sx2, const float* __restrict__ se2,
    const float* __restrict__ dv2e, const float* __restrict__ gv2e,
    const float* __restrict__ de2v, const float* __restrict__ ge2v,
    const float* __restrict__ attn_b, const float* __restrict__ wd_w,
    const float* __restrict__ wd_b, const float* __restrict__ wg_w,
    const float* __restrict__ wg_b, int layer,
    float* sV2E, float* sE2V, unsigned* mEdgeU, unsigned* mNodeU, int nnz) {
  int k = blockIdx.x * 256 + threadIdx.x;
  if (k >= nnz) return;
  float ab  = attn_b[layer];
  float wdw = wd_w[layer], wdb = wd_b[layer];
  float wgw = wg_w[layer], wgb = wg_b[layer];
  int n = nodes[k], e = edges[k];
  float base = ab + wdb + wgb;
  float sV = sx1[n] + se1[e] + base + dv2e[k] * wdw + gv2e[k] * wgw;
  sV2E[k] = sV;
  atomicMax(mEdgeU + e, f2key(sV));
  float sE = se2[e] + sx2[n] + base + de2v[k] * wdw + ge2v[k] * wgw;
  sE2V[k] = sE;
  atomicMax(mNodeU + n, f2key(sE));
}

// ---------- exp + segment denominators ----------
__global__ void softden_kernel(
    const int* __restrict__ nodes, const int* __restrict__ edges,
    const float* __restrict__ sV2E, const float* __restrict__ sE2V,
    const unsigned* __restrict__ mEdgeU, const unsigned* __restrict__ mNodeU,
    float* exV2E, float* exE2V, float* denEdge, float* denNode, int nnz) {
  int k = blockIdx.x * 256 + threadIdx.x;
  if (k >= nnz) return;
  int n = nodes[k], e = edges[k];
  float exv = __expf(sV2E[k] - key2f(mEdgeU[e]));
  exV2E[k] = exv;
  atomicAdd(denEdge + e, exv);
  float exe = __expf(sE2V[k] - key2f(mNodeU[n]));
  exE2V[k] = exe;
  atomicAdd(denNode + n, exe);
}

// ---------- scatter pass 1: XeA += x[n]*aV (per edge), XvA += e[e]*aE (per node) ----------
__global__ __launch_bounds__(256) void scatter1_kernel(
    const int* __restrict__ nodes, const int* __restrict__ edges,
    const float* __restrict__ xcur, const float* __restrict__ ecur,
    const float* __restrict__ exV, const float* __restrict__ exE,
    const float* __restrict__ denE, const float* __restrict__ denN,
    float* XeA, float* XvA, int nnz) {
  int k = blockIdx.x * 8 + (threadIdx.x >> 5);
  if (k >= nnz) return;
  int lane = threadIdx.x & 31;
  int d0 = lane * 4;
  int n = nodes[k], e = edges[k];
  float aV = exV[k] / denE[e];
  float aE = exE[k] / denN[n];
  float4 xv = *(const float4*)(xcur + (size_t)n * DD + d0);
  float* pe = XeA + (size_t)e * DD + d0;
  atomicAdd(pe + 0, xv.x * aV);
  atomicAdd(pe + 1, xv.y * aV);
  atomicAdd(pe + 2, xv.z * aV);
  atomicAdd(pe + 3, xv.w * aV);
  float4 ev = *(const float4*)(ecur + (size_t)e * DD + d0);
  float* pn = XvA + (size_t)n * DD + d0;
  atomicAdd(pn + 0, ev.x * aE);
  atomicAdd(pn + 1, ev.y * aE);
  atomicAdd(pn + 2, ev.z * aE);
  atomicAdd(pn + 3, ev.w * aE);
}

// ---------- scatter pass 2: XvB += (XeA*degE)[e]*aE ; XeB += (XvA*degV)[n]*aV ----------
__global__ __launch_bounds__(256) void scatter2_kernel(
    const int* __restrict__ nodes, const int* __restrict__ edges,
    const float* __restrict__ exV, const float* __restrict__ exE,
    const float* __restrict__ denE, const float* __restrict__ denN,
    const float* __restrict__ degE, const float* __restrict__ degV,
    const float* __restrict__ XeA, const float* __restrict__ XvA,
    float* XeB, float* XvB, int nnz) {
  int k = blockIdx.x * 8 + (threadIdx.x >> 5);
  if (k >= nnz) return;
  int lane = threadIdx.x & 31;
  int d0 = lane * 4;
  int n = nodes[k], e = edges[k];
  float aV = exV[k] / denE[e];
  float aE = exE[k] / denN[n];
  float sE = degE[e] * aE;   // scale for Xe1[e] -> node scatter
  float sV = degV[n] * aV;   // scale for Xv2[n] -> edge scatter
  float4 xe = *(const float4*)(XeA + (size_t)e * DD + d0);
  float* pn = XvB + (size_t)n * DD + d0;
  atomicAdd(pn + 0, xe.x * sE);
  atomicAdd(pn + 1, xe.y * sE);
  atomicAdd(pn + 2, xe.z * sE);
  atomicAdd(pn + 3, xe.w * sE);
  float4 xv = *(const float4*)(XvA + (size_t)n * DD + d0);
  float* pe = XeB + (size_t)e * DD + d0;
  atomicAdd(pe + 0, xv.x * sV);
  atomicAdd(pe + 1, xv.y * sV);
  atomicAdd(pe + 2, xv.z * sV);
  atomicAdd(pe + 3, xv.w * sV);
}

// ---------- WMMA GEMM: out[M,128] from A[M,128] @ W[128,128] ----------
// WT is the transposed bf16 weight: WT[n*128 + k] = W[k][n].
// MODE 0: A read directly, out = relu(A@W + bias)
// MODE 1: A = xi = (1-alpha)*0.5*deg[row]*(S1+S2) + alpha*X0,
//         out = (1-beta)*xi + beta*(xi@W), optional relu
// REQUIRES: M is a multiple of 16 (true for 100000 and 20000).
template <int MODE>
__global__ __launch_bounds__(256) void gemm128_kernel(
    const float* __restrict__ A, const float* __restrict__ S1,
    const float* __restrict__ S2, const float* __restrict__ X0,
    const float* __restrict__ deg, const bf16_t* __restrict__ WT,
    const float* __restrict__ bias, float* __restrict__ out,
    int do_relu, float alpha, float beta) {
  // A tile stored in WMMA-fragment order: AF[(ks*32 + lane)*16 + j]
  __shared__ __align__(32) bf16_t AF[16 * DD];   // 4 KB
  __shared__ float Xs[16 * DD];                  // 8 KB, xi f32 for epilogue

  const int tid = threadIdx.x;
  const int r0  = blockIdx.x * 16;

  // stage 16x128 A tile: f32 -> Xs, bf16 -> AF (fragment-swizzled, packed quads)
  for (int t = tid * 4; t < 16 * DD; t += 256 * 4) {
    int m = t >> 7, k = t & 127;
    int row = r0 + m;
    float4 v;
    if (MODE == 1) {
      float4 a1 = *(const float4*)(S1 + (size_t)row * DD + k);
      float4 a2 = *(const float4*)(S2 + (size_t)row * DD + k);
      float4 a0 = *(const float4*)(X0 + (size_t)row * DD + k);
      float c1 = (1.0f - alpha) * 0.5f * deg[row];
      v.x = c1 * (a1.x + a2.x) + alpha * a0.x;
      v.y = c1 * (a1.y + a2.y) + alpha * a0.y;
      v.z = c1 * (a1.z + a2.z) + alpha * a0.z;
      v.w = c1 * (a1.w + a2.w) + alpha * a0.w;
    } else {
      v = *(const float4*)(A + (size_t)row * DD + k);
    }
    Xs[m * DD + k + 0] = v.x; Xs[m * DD + k + 1] = v.y;
    Xs[m * DD + k + 2] = v.z; Xs[m * DD + k + 3] = v.w;
    // contiguous-k (k = 16*half + j) A-layout inverse: any aligned k-quad maps to 4 consecutive j of one lane
    int ks   = k >> 5;
    int rem  = k & 31;
    int half = rem >> 4;                     // K 16-31 live in lanes 16-31
    int w_   = rem - 16 * half;              // in {0..15}
    int j0   = w_;                           // fragment element index
    bf16_t* dp = &AF[((ks * 32) + half * 16 + m) * 16 + j0];
    dp[0] = (bf16_t)v.x; dp[1] = (bf16_t)v.y;
    dp[2] = (bf16_t)v.z; dp[3] = (bf16_t)v.w;
  }
  __syncthreads();

  const int wave = tid >> 5;
  const int lane = tid & 31;
  const int half = lane >> 4;
  const int lm   = lane & 15;
  const int n0   = wave * 16;        // each wave owns one 16-wide N tile

  const v16bf* Av = (const v16bf*)AF;
  const v16bf* Bv = (const v16bf*)WT;   // 16-elem granules, 32B aligned
  const int brow = (n0 + lm) * 8;       // (n)*128/16 granules per row

  v8f c = {0.f, 0.f, 0.f, 0.f, 0.f, 0.f, 0.f, 0.f};
#pragma unroll
  for (int ks = 0; ks < 4; ++ks) {      // K = 128 = 4 x 32
    v16bf a = Av[ks * 32 + lane];                 // contiguous 32B LDS load
    v16bf b = Bv[brow + ks * 2 + half];           // contiguous 32B global load
    c = __builtin_amdgcn_wmma_f32_16x16x32_bf16(false, a, false, b,
                                                (short)0, c, false, false);
  }

  const int col = n0 + lm;
  const float bcol = (MODE == 0) ? bias[col] : 0.0f;
#pragma unroll
  for (int r = 0; r < 8; ++r) {
    int m = r + 8 * half;   // C/D layout: VGPR r -> M=r (lanes 0-15), M=8+r (16-31)
    int row = r0 + m;
    float acc = c[r];
    float o;
    if (MODE == 1) {
      float xi = Xs[m * DD + col];
      o = (1.0f - beta) * xi + beta * acc;
    } else {
      o = acc + bcol;
    }
    if (do_relu) o = fmaxf(o, 0.0f);
    out[(size_t)row * DD + col] = o;
  }
}

// =======================================================================
extern "C" void kernel_launch(void* const* d_in, const int* in_sizes, int n_in,
                              void* d_out, int out_size, void* d_ws, size_t ws_size,
                              hipStream_t stream) {
  (void)in_sizes; (void)n_in; (void)out_size; (void)ws_size;
  const float* x_in  = (const float*)d_in[0];
  const float* e_in  = (const float*)d_in[1];
  const int*   hei   = (const int*)d_in[2];
  const int*   nodes = hei;
  const int*   edges = hei + NNZV;
  const float* dv2e  = (const float*)d_in[3];
  const float* gv2e  = (const float*)d_in[4];
  const float* de2v  = (const float*)d_in[5];
  const float* ge2v  = (const float*)d_in[6];
  const float* W0    = (const float*)d_in[7];
  const float* b0    = (const float*)d_in[8];
  const float* Wn    = (const float*)d_in[9];
  const float* We    = (const float*)d_in[10];
  const float* aw    = (const float*)d_in[11];
  const float* ab    = (const float*)d_in[12];
  const float* wdw   = (const float*)d_in[13];
  const float* wdb   = (const float*)d_in[14];
  const float* wgw   = (const float*)d_in[15];
  const float* wgb   = (const float*)d_in[16];

  // ---- workspace carve-up ----
  float* ws = (float*)d_ws;
  size_t o = 0;
  // 9 transposed bf16 weight matrices (W0, Wn[0..3], We[0..3]): 9*16384 bf16
  bf16_t* WTall = (bf16_t*)ws;
  o += (9 * DD * DD + 1) / 2;          // bf16 pairs per float slot
  o = (o + 7) & ~(size_t)7;            // keep later buffers 32B aligned
  auto alloc = [&](size_t n) { float* p = ws + o; o += n; return p; };
  const size_t ND = (size_t)N_NODES * DD, ED = (size_t)N_EDGES * DD;
  float* xA   = alloc(ND);           // x0 = relu(x@W0+b0), also layer-0 input
  float* xB   = alloc(ND);
  float* xC   = alloc(ND);
  float* eB   = alloc(ED);
  float* eC   = alloc(ED);
  float* XvA  = alloc(ND);           // raw Xv2 accumulator
  float* XvB  = alloc(ND);           // raw Xv1 accumulator (adjacent)
  float* XeA  = alloc(ED);           // raw Xe1 accumulator
  float* XeB  = alloc(ED);           // raw Xe2 accumulator (adjacent)
  float* mNode   = alloc(N_NODES);   // ordered-uint max keys
  float* denNode = alloc(N_NODES);
  float* mEdge   = alloc(N_EDGES);
  float* denEdge = alloc(N_EDGES);
  float* dV    = alloc(N_NODES);
  float* cntE  = alloc(N_EDGES);
  float* dEsum = alloc(N_EDGES);     // adjacent to cntE
  float* degV  = alloc(N_NODES);
  float* degE  = alloc(N_EDGES);
  float* sx1 = alloc(N_NODES); float* sx2 = alloc(N_NODES);
  float* se1 = alloc(N_EDGES); float* se2 = alloc(N_EDGES);
  float* sV2E  = alloc(NNZV); float* sE2V  = alloc(NNZV);
  float* exV2E = alloc(NNZV); float* exE2V = alloc(NNZV);

  auto fill0 = [&](float* p, size_t n) {
    fill0_kernel<<<dim3((unsigned)((n + 255) / 256)), dim3(256), 0, stream>>>(p, (int)n);
  };
  const unsigned gNNZ  = (NNZV + 255) / 256;
  const unsigned gNNZ8 = (NNZV + 7) / 8;

  // ---- weights: transpose + convert to bf16 once ----
  wtrans_kernel<<<(9 * DD * DD + 255) / 256, 256, 0, stream>>>(W0, Wn, We, WTall);

  // ---- degrees ----
  fill0(dV, N_NODES);
  fill0(cntE, 2 * (size_t)N_EDGES);  // cntE + dEsum
  deg_count_kernel<<<gNNZ, 256, 0, stream>>>(nodes, edges, dV, cntE, NNZV);
  deg_sum_kernel<<<gNNZ, 256, 0, stream>>>(nodes, edges, dV, dEsum, NNZV);
  deg_final_kernel<<<(N_NODES + 255) / 256, 256, 0, stream>>>(
      dV, cntE, dEsum, degV, degE, N_NODES, N_EDGES);

  // ---- initial x = relu(x @ W0 + b0) ----
  gemm128_kernel<0><<<N_NODES / 16, 256, 0, stream>>>(
      x_in, nullptr, nullptr, nullptr, nullptr, WTall, b0, xA,
      /*relu=*/1, 0.0f, 0.0f);

  float* outx = (float*)d_out;
  float* oute = outx + ND;

  const float* xcur = xA;
  const float* ecur = e_in;
  for (int i = 0; i < NLAYERS; ++i) {
    float beta = logf(LAMDA_C / (float)(i + 1) + 1.0f);

    fill0(XvA, 2 * ND);                       // XvA + XvB
    fill0(XeA, 2 * ED);                       // XeA + XeB
    fill0(mNode, 2 * (size_t)N_NODES);        // mNode + denNode (key 0 == -inf)
    fill0(mEdge, 2 * (size_t)N_EDGES);        // mEdge + denEdge

    proj_kernel<<<(N_NODES + N_EDGES + 7) / 8, 256, 0, stream>>>(
        xcur, ecur, aw, i, sx1, sx2, se1, se2, N_NODES, N_EDGES);

    scores_kernel<<<gNNZ, 256, 0, stream>>>(
        nodes, edges, sx1, se1, sx2, se2, dv2e, gv2e, de2v, ge2v,
        ab, wdw, wdb, wgw, wgb, i,
        sV2E, sE2V, (unsigned*)mEdge, (unsigned*)mNode, NNZV);

    softden_kernel<<<gNNZ, 256, 0, stream>>>(
        nodes, edges, sV2E, sE2V, (const unsigned*)mEdge, (const unsigned*)mNode,
        exV2E, exE2V, denEdge, denNode, NNZV);

    scatter1_kernel<<<gNNZ8, 256, 0, stream>>>(
        nodes, edges, xcur, ecur, exV2E, exE2V, denEdge, denNode,
        XeA, XvA, NNZV);

    scatter2_kernel<<<gNNZ8, 256, 0, stream>>>(
        nodes, edges, exV2E, exE2V, denEdge, denNode, degE, degV,
        XeA, XvA, XeB, XvB, NNZV);

    float* xout = (i == 0) ? xB : (i == 1) ? xC : (i == 2) ? xB : outx;
    float* eout = (i == 0) ? eB : (i == 1) ? eC : (i == 2) ? eB : oute;
    int relu = (i < NLAYERS - 1) ? 1 : 0;

    gemm128_kernel<1><<<N_NODES / 16, 256, 0, stream>>>(
        nullptr, XvA, XvB, xA, degV, WTall + (size_t)(1 + i) * DD * DD, nullptr,
        xout, relu, ALPHA_C, beta);
    gemm128_kernel<1><<<N_EDGES / 16, 256, 0, stream>>>(
        nullptr, XeA, XeB, e_in, degE, WTall + (size_t)(5 + i) * DD * DD, nullptr,
        eout, relu, ALPHA_C, beta);

    xcur = xout;
    ecur = eout;
  }
}
